// RNNModel_1640677507197
// MI455X (gfx1250) — compile-verified, matrix operand order checked by reading, not run on hardware
//
#include <hip/hip_runtime.h>
#include <cstdint>

typedef __attribute__((ext_vector_type(16))) __bf16 v16bf;
typedef __attribute__((ext_vector_type(8)))  float  v8f;
typedef unsigned int u32x4 __attribute__((ext_vector_type(4)));
typedef int          i32x8 __attribute__((ext_vector_type(8)));
typedef int          i32x4 __attribute__((ext_vector_type(4)));

#define DEV __device__ __forceinline__

constexpr int Bsz = 64, Tseq = 512, Iin = 256, Hdim = 512, Odim = 10;
constexpr int G4 = 4 * Hdim;   // 2048 (lstm gates)
constexpr int G3 = 3 * Hdim;   // 1536 (gru gates)
constexpr size_t Mrows = (size_t)Bsz * Tseq;   // 32768

// ---- WMMA 16x16x32 bf16 fragment index math (CDNA5 ISA, 16-bit A layout) ----
DEV int frag_k(int hi, int e) {
  int v = e >> 1, p = e & 1;
  return (v < 4) ? (hi * 8 + 2 * v + p) : (16 + hi * 8 + 2 * (v - 4) + p);
}
DEV void frag_inv(int k, int& hi, int& e) {
  hi = (k >> 3) & 1;
  e  = (k & 7) + ((k >> 4) << 3);
}

DEV v8f wmma_bf16(v16bf a, v16bf b, v8f c) {
  return __builtin_amdgcn_wmma_f32_16x16x32_bf16(false, a, false, b, (short)0, c,
                                                 false, false);
}

DEV float sigm(float x)  { return 1.0f / (1.0f + __expf(-x)); }
DEV float tanh_(float x) { return 2.0f / (1.0f + __expf(-2.0f * x)) - 1.0f; }

// atomic grid barrier: 16 resident workgroups, monotonic counter per scan
DEV void grid_sync(unsigned* cnt, unsigned target) {
  __threadfence();
  __syncthreads();
  if (threadIdx.x == 0) {
    atomicAdd(cnt, 1u);
    while (atomicAdd(cnt, 0u) < target) __builtin_amdgcn_s_sleep(2);
  }
  __syncthreads();
  __threadfence();
}

// ---- Tensor Data Mover: 2D tile Global -> LDS (D# per CDNA5 ISA) ----
// d0_units: contiguous run length in 8-byte units; d1_rows: number of rows;
// row_stride_units: stride between rows in 8-byte units.
DEV void tdm_load_2d(const void* gsrc, unsigned lds_off_bytes,
                     unsigned d0_units, unsigned d1_rows,
                     unsigned long long row_stride_units) {
  unsigned long long ga = (unsigned long long)(uintptr_t)gsrc;
  u32x4 g0;
  g0.x = 1u;                                             // count=1, user mode
  g0.y = lds_off_bytes;                                  // lds_addr (bytes)
  g0.z = (unsigned)ga;                                   // global_addr[31:0]
  g0.w = (unsigned)((ga >> 32) & 0x1FFFFFFull) | (2u << 30); // ga[56:32], type=2
  i32x8 g1;
  g1[0] = (int)(3u << 16);                               // data_size=3 (8B units)
  g1[1] = (int)((d0_units & 0xFFFFu) << 16);             // tensor_dim0[15:0]
  g1[2] = (int)((d0_units >> 16) | ((d1_rows & 0xFFFFu) << 16)); // td0 hi | td1 lo
  g1[3] = (int)((d1_rows >> 16) | ((d0_units & 0xFFFFu) << 16)); // td1 hi | tile_dim0
  g1[4] = (int)(d1_rows & 0xFFFFu);                      // tile_dim1 (tile_dim2=0)
  g1[5] = (int)(unsigned)(row_stride_units & 0xFFFFFFFFull); // dim0_stride lo32
  g1[6] = (int)(unsigned)((row_stride_units >> 32) & 0xFFFFull); // dim0_stride hi16
  g1[7] = 0;
  i32x4 gz4 = {0, 0, 0, 0};
  i32x8 gz8 = {0, 0, 0, 0, 0, 0, 0, 0};
  // 6-arg form (this toolchain): (g0, g1, g2, g3, g4, cpol)
  __builtin_amdgcn_tensor_load_to_lds(g0, g1, gz4, gz4, gz8, 0);
}

extern __shared__ __bf16 smem[];   // dynamic LDS: staged recurrent weights

// ---- repack row-major (rows x K) fp32/bf16 -> bf16 WMMA fragment order ----
template <typename SRC>
__global__ __launch_bounds__(256) void repack_frag(const SRC* __restrict__ src,
                                                   __bf16* __restrict__ dst,
                                                   int rows, int K) {
  size_t idx = (size_t)blockIdx.x * 256 + threadIdx.x;
  if (idx >= (size_t)rows * (size_t)K) return;
  int e    = (int)(idx & 15);
  int lane = (int)((idx >> 4) & 31);
  size_t rest = idx >> 9;
  int KT = K >> 5;
  int kt = (int)(rest % (size_t)KT);
  int rt = (int)(rest / (size_t)KT);
  int row = rt * 16 + (lane & 15);
  int k   = kt * 32 + frag_k(lane >> 4, e);
  dst[idx] = (__bf16)(float)src[(size_t)row * K + k];
}

// ---- bf16 GEMM: out(T,B,N) = A(M,K) * B(N,K)^T + bias, one wave per 16x16 tile ----
__global__ __launch_bounds__(256) void gemm_bf16(const __bf16* __restrict__ Af,
                                                 const __bf16* __restrict__ Bf,
                                                 float* __restrict__ out,
                                                 const float* __restrict__ ba,
                                                 const float* __restrict__ bbias,
                                                 int Mtiles, int Ntiles, int KT,
                                                 int N) {
  int waveg = blockIdx.x * 8 + (threadIdx.x >> 5);
  int lane  = threadIdx.x & 31;
  int nt = waveg % Ntiles;
  int mt = waveg / Ntiles;
  if (mt >= Mtiles) return;   // wave-uniform: EXEC stays all-ones for WMMA
  const __bf16* ap = Af + (size_t)mt * KT * 512 + (size_t)lane * 16;
  const __bf16* bp = Bf + (size_t)nt * KT * 512 + (size_t)lane * 16;
  v8f acc = {};
  for (int kt = 0; kt < KT; ++kt) {
    v16bf a = *(const v16bf*)ap;
    v16bf b = *(const v16bf*)bp;
    acc = wmma_bf16(a, b, acc);
    ap += 512; bp += 512;
  }
  int col = nt * 16 + (lane & 15);
  float bias = 0.0f;
  if (ba)    bias += ba[col];
  if (bbias) bias += bbias[col];
  int hi = lane >> 4;
#pragma unroll
  for (int r = 0; r < 8; ++r) {
    int m  = mt * 16 + hi * 8 + r;      // source row = b*T + t
    int b_ = m >> 9;                    // / Tseq (512)
    int t_ = m & (Tseq - 1);
    out[((size_t)t_ * Bsz + b_) * N + col] = acc[r] + bias;
  }
}

// zero h fragment double-buffer + barrier counter
__global__ __launch_bounds__(256) void scan_init(__bf16* hf0, __bf16* hf1, unsigned* cnt) {
  size_t idx = (size_t)blockIdx.x * 256 + threadIdx.x;
  if (idx == 0) *cnt = 0u;
  if (idx < (size_t)Bsz * Hdim) {
    hf0[idx] = (__bf16)0.0f;
    hf1[idx] = (__bf16)0.0f;
  }
}

// ---- LSTM scan: 16 WGs, each owns 32 H-columns; c kept in registers; weights in LDS ----
__global__ __launch_bounds__(256) void lstm_scan(const float* __restrict__ xg,   // (T,B,4H), biases folded
                                                 const __bf16* __restrict__ whh, // frag, N=2048 K=512
                                                 __bf16* __restrict__ hf0,
                                                 __bf16* __restrict__ hf1,
                                                 __bf16* __restrict__ Hout,      // (B,T,H) bf16
                                                 unsigned* __restrict__ cnt) {
  const int wg   = blockIdx.x;          // 0..15
  const int wave = threadIdx.x >> 5;    // 0..7  -> (mt, ct)
  const int lane = threadIdx.x & 31;
  const int mt = wave >> 1, ct = wave & 1;
  const int hi = lane >> 4;
  const int n0   = wg * 32 + ct * 16;
  const int hcol = n0 + (lane & 15);    // this lane's H column
  constexpr int KT = Hdim / 32;         // 16

  // Stage this WG's w_hh slice into LDS via TDM:
  // 4 gate rows x 32KB contiguous (both ct tiles), row stride = 32 tiles = 512KB
  if (wave == 0) {
    tdm_load_2d(whh + (size_t)(wg * 2) * KT * 512,
                (unsigned)(uintptr_t)smem,
                /*d0_units=*/4096u, /*d1_rows=*/4u,
                /*row_stride_units=*/65536ull);
    __builtin_amdgcn_s_wait_tensorcnt(0);
  }
  __syncthreads();

  float c[8];
#pragma unroll
  for (int r = 0; r < 8; ++r) c[r] = 0.0f;

  int fhi, fe;
  frag_inv(hcol & 31, fhi, fe);
  const int fkt = hcol >> 5;

  // LDS weight bases: [g][ct][kt][lane][e]
  const __bf16* wb0 = smem + (size_t)((0 * 2 + ct) * KT) * 512 + (size_t)lane * 16;
  const __bf16* wb1 = smem + (size_t)((1 * 2 + ct) * KT) * 512 + (size_t)lane * 16;
  const __bf16* wb2 = smem + (size_t)((2 * 2 + ct) * KT) * 512 + (size_t)lane * 16;
  const __bf16* wb3 = smem + (size_t)((3 * 2 + ct) * KT) * 512 + (size_t)lane * 16;

#pragma unroll 1
  for (int t = 0; t < Tseq; ++t) {
    const __bf16* hcur = (t & 1) ? hf1 : hf0;
    __bf16*       hnxt = (t & 1) ? hf0 : hf1;

    v8f a0 = {}, a1 = {}, a2 = {}, a3 = {};
    const float* xp = xg + (size_t)t * Bsz * G4;
#pragma unroll
    for (int r = 0; r < 8; ++r) {
      int row = mt * 16 + hi * 8 + r;
      const float* p = xp + (size_t)row * G4 + hcol;
      a0[r] = p[0];
      a1[r] = p[Hdim];
      a2[r] = p[2 * Hdim];
      a3[r] = p[3 * Hdim];
    }

    const __bf16* ap = hcur + (size_t)mt * KT * 512 + (size_t)lane * 16;
    const __bf16* b0 = wb0;
    const __bf16* b1 = wb1;
    const __bf16* b2 = wb2;
    const __bf16* b3 = wb3;
    for (int kt = 0; kt < KT; ++kt) {
      v16bf a = *(const v16bf*)ap;
      a0 = wmma_bf16(a, *(const v16bf*)b0, a0);
      a1 = wmma_bf16(a, *(const v16bf*)b1, a1);
      a2 = wmma_bf16(a, *(const v16bf*)b2, a2);
      a3 = wmma_bf16(a, *(const v16bf*)b3, a3);
      ap += 512; b0 += 512; b1 += 512; b2 += 512; b3 += 512;
    }

#pragma unroll
    for (int r = 0; r < 8; ++r) {
      float ig = sigm(a0[r]);
      float fg = sigm(a1[r]);
      float gg = tanh_(a2[r]);
      float og = sigm(a3[r]);
      c[r] = fg * c[r] + ig * gg;
      float h = og * tanh_(c[r]);
      __bf16 hb = (__bf16)h;
      int row = mt * 16 + hi * 8 + r;           // batch index
      Hout[((size_t)row * Tseq + t) * Hdim + hcol] = hb;
      int flane = fhi * 16 + (row & 15);
      hnxt[(((size_t)mt * KT + fkt) * 32 + flane) * 16 + fe] = hb;
    }
    grid_sync(cnt, 16u * (unsigned)(t + 1));
  }
}

// ---- GRU scan ----
__global__ __launch_bounds__(256) void gru_scan(const float* __restrict__ xg,   // (T,B,3H), b_ih folded
                                                const __bf16* __restrict__ whh, // frag, N=1536 K=512
                                                const float* __restrict__ bhh,
                                                __bf16* __restrict__ hf0,
                                                __bf16* __restrict__ hf1,
                                                __bf16* __restrict__ Hout,      // (B,T,H) bf16
                                                unsigned* __restrict__ cnt) {
  const int wg   = blockIdx.x;
  const int wave = threadIdx.x >> 5;
  const int lane = threadIdx.x & 31;
  const int mt = wave >> 1, ct = wave & 1;
  const int hi = lane >> 4;
  const int n0   = wg * 32 + ct * 16;
  const int hcol = n0 + (lane & 15);
  constexpr int KT = Hdim / 32;

  if (wave == 0) {
    tdm_load_2d(whh + (size_t)(wg * 2) * KT * 512,
                (unsigned)(uintptr_t)smem,
                4096u, 3u, 65536ull);
    __builtin_amdgcn_s_wait_tensorcnt(0);
  }
  __syncthreads();

  float hreg[8];
#pragma unroll
  for (int r = 0; r < 8; ++r) hreg[r] = 0.0f;

  int fhi, fe;
  frag_inv(hcol & 31, fhi, fe);
  const int fkt = hcol >> 5;

  const float bh_r = bhh[hcol];
  const float bh_z = bhh[Hdim + hcol];
  const float bh_n = bhh[2 * Hdim + hcol];

  const __bf16* wb0 = smem + (size_t)((0 * 2 + ct) * KT) * 512 + (size_t)lane * 16;
  const __bf16* wb1 = smem + (size_t)((1 * 2 + ct) * KT) * 512 + (size_t)lane * 16;
  const __bf16* wb2 = smem + (size_t)((2 * 2 + ct) * KT) * 512 + (size_t)lane * 16;

#pragma unroll 1
  for (int t = 0; t < Tseq; ++t) {
    const __bf16* hcur = (t & 1) ? hf1 : hf0;
    __bf16*       hnxt = (t & 1) ? hf0 : hf1;

    v8f ar = {}, az = {}, an = {};
#pragma unroll
    for (int r = 0; r < 8; ++r) { ar[r] = bh_r; az[r] = bh_z; an[r] = bh_n; }

    const __bf16* ap = hcur + (size_t)mt * KT * 512 + (size_t)lane * 16;
    const __bf16* b0 = wb0;
    const __bf16* b1 = wb1;
    const __bf16* b2 = wb2;
    for (int kt = 0; kt < KT; ++kt) {
      v16bf a = *(const v16bf*)ap;
      ar = wmma_bf16(a, *(const v16bf*)b0, ar);
      az = wmma_bf16(a, *(const v16bf*)b1, az);
      an = wmma_bf16(a, *(const v16bf*)b2, an);
      ap += 512; b0 += 512; b1 += 512; b2 += 512;
    }

    const float* xp = xg + (size_t)t * Bsz * G3;
#pragma unroll
    for (int r = 0; r < 8; ++r) {
      int row = mt * 16 + hi * 8 + r;
      const float* p = xp + (size_t)row * G3 + hcol;
      float rg = sigm(p[0] + ar[r]);
      float zg = sigm(p[Hdim] + az[r]);
      float ng = tanh_(p[2 * Hdim] + rg * an[r]);
      float h = (1.0f - zg) * ng + zg * hreg[r];
      hreg[r] = h;
      __bf16 hb = (__bf16)h;
      Hout[((size_t)row * Tseq + t) * Hdim + hcol] = hb;
      int flane = fhi * 16 + (row & 15);
      hnxt[(((size_t)mt * KT + fkt) * 32 + flane) * 16 + fe] = hb;
    }
    grid_sync(cnt, 16u * (unsigned)(t + 1));
  }
}

// ---- RNN-tanh scan (keeps only final h, fp32) ----
__global__ __launch_bounds__(256) void rnn_scan(const float* __restrict__ xg,   // (T,B,H), biases folded
                                                const __bf16* __restrict__ whh, // frag, N=512 K=512
                                                __bf16* __restrict__ hf0,
                                                __bf16* __restrict__ hf1,
                                                float* __restrict__ hlast,      // (B,H) fp32
                                                unsigned* __restrict__ cnt) {
  const int wg   = blockIdx.x;
  const int wave = threadIdx.x >> 5;
  const int lane = threadIdx.x & 31;
  const int mt = wave >> 1, ct = wave & 1;
  const int hi = lane >> 4;
  const int n0   = wg * 32 + ct * 16;
  const int hcol = n0 + (lane & 15);
  constexpr int KT = Hdim / 32;

  if (wave == 0) {
    tdm_load_2d(whh + (size_t)(wg * 2) * KT * 512,
                (unsigned)(uintptr_t)smem,
                4096u, 1u, 4096ull);
    __builtin_amdgcn_s_wait_tensorcnt(0);
  }
  __syncthreads();

  int fhi, fe;
  frag_inv(hcol & 31, fhi, fe);
  const int fkt = hcol >> 5;

  const __bf16* wb = smem + (size_t)(ct * KT) * 512 + (size_t)lane * 16;

#pragma unroll 1
  for (int t = 0; t < Tseq; ++t) {
    const __bf16* hcur = (t & 1) ? hf1 : hf0;
    __bf16*       hnxt = (t & 1) ? hf0 : hf1;

    v8f acc = {};
    const float* xp = xg + (size_t)t * Bsz * Hdim;
#pragma unroll
    for (int r = 0; r < 8; ++r) {
      int row = mt * 16 + hi * 8 + r;
      acc[r] = xp[(size_t)row * Hdim + hcol];
    }

    const __bf16* ap = hcur + (size_t)mt * KT * 512 + (size_t)lane * 16;
    const __bf16* bp = wb;
    for (int kt = 0; kt < KT; ++kt) {
      v16bf a = *(const v16bf*)ap;
      acc = wmma_bf16(a, *(const v16bf*)bp, acc);
      ap += 512; bp += 512;
    }

#pragma unroll
    for (int r = 0; r < 8; ++r) {
      float h = tanh_(acc[r]);
      int row = mt * 16 + hi * 8 + r;
      int flane = fhi * 16 + (row & 15);
      hnxt[(((size_t)mt * KT + fkt) * 32 + flane) * 16 + fe] = (__bf16)h;
      if (t == Tseq - 1) hlast[(size_t)row * Hdim + hcol] = h;
    }
    grid_sync(cnt, 16u * (unsigned)(t + 1));
  }
}

// ---- FC head: (64,512) x (10,512)^T + b ----
__global__ __launch_bounds__(256) void fc_kernel(const float* __restrict__ hlast,
                                                 const float* __restrict__ w,
                                                 const float* __restrict__ bias,
                                                 float* __restrict__ out) {
  int idx = blockIdx.x * 256 + threadIdx.x;
  if (idx >= Bsz * Odim) return;
  int b = idx / Odim;
  int o = idx - b * Odim;
  float s = bias[o];
  for (int n = 0; n < Hdim; ++n) s += hlast[(size_t)b * Hdim + n] * w[(size_t)o * Hdim + n];
  out[idx] = s;
}

extern "C" void kernel_launch(void* const* d_in, const int* in_sizes, int n_in,
                              void* d_out, int out_size, void* d_ws, size_t ws_size,
                              hipStream_t stream) {
  (void)in_sizes; (void)n_in; (void)out_size; (void)ws_size;
  const float* x        = (const float*)d_in[0];
  const float* lstm_wih = (const float*)d_in[1];
  const float* lstm_whh = (const float*)d_in[2];
  const float* lstm_bih = (const float*)d_in[3];
  const float* lstm_bhh = (const float*)d_in[4];
  const float* gru_wih  = (const float*)d_in[5];
  const float* gru_whh  = (const float*)d_in[6];
  const float* gru_bih  = (const float*)d_in[7];
  const float* gru_bhh  = (const float*)d_in[8];
  const float* rnn_wih  = (const float*)d_in[9];
  const float* rnn_whh  = (const float*)d_in[10];
  const float* rnn_bih  = (const float*)d_in[11];
  const float* rnn_bhh  = (const float*)d_in[12];
  const float* fc_w     = (const float*)d_in[13];
  const float* fc_b     = (const float*)d_in[14];

  char* wsp = (char*)d_ws;
  auto take = [&](size_t bytes) {
    char* p = wsp;
    wsp += (bytes + 255) & ~(size_t)255;
    return p;
  };
  __bf16* xfrag = (__bf16*)take(Mrows * Iin * 2);
  __bf16* wihL  = (__bf16*)take((size_t)G4 * Iin * 2);
  __bf16* whhL  = (__bf16*)take((size_t)G4 * Hdim * 2);
  __bf16* wihG  = (__bf16*)take((size_t)G3 * Hdim * 2);
  __bf16* whhG  = (__bf16*)take((size_t)G3 * Hdim * 2);
  __bf16* wihR  = (__bf16*)take((size_t)Hdim * Hdim * 2);
  __bf16* whhR  = (__bf16*)take((size_t)Hdim * Hdim * 2);
  float*  xgbuf = (float*)take((size_t)Tseq * Bsz * G4 * 4);   // reused for all 3 layers
  __bf16* Hrow  = (__bf16*)take(Mrows * Hdim * 2);             // H1, then H2 (row-major)
  __bf16* Hfrag = (__bf16*)take(Mrows * Hdim * 2);             // repacked A fragments
  __bf16* hf0   = (__bf16*)take((size_t)Bsz * Hdim * 2);
  __bf16* hf1   = (__bf16*)take((size_t)Bsz * Hdim * 2);
  float*  hlast = (float*)take((size_t)Bsz * Hdim * 4);
  unsigned* cnt = (unsigned*)take(256);

  auto nblk = [](size_t total) { return (unsigned)((total + 255) / 256); };

  // weight / activation repacks into WMMA fragment order (bf16)
  repack_frag<float><<<nblk(Mrows * Iin), 256, 0, stream>>>(x, xfrag, (int)Mrows, Iin);
  repack_frag<float><<<nblk((size_t)G4 * Iin), 256, 0, stream>>>(lstm_wih, wihL, G4, Iin);
  repack_frag<float><<<nblk((size_t)G4 * Hdim), 256, 0, stream>>>(lstm_whh, whhL, G4, Hdim);
  repack_frag<float><<<nblk((size_t)G3 * Hdim), 256, 0, stream>>>(gru_wih, wihG, G3, Hdim);
  repack_frag<float><<<nblk((size_t)G3 * Hdim), 256, 0, stream>>>(gru_whh, whhG, G3, Hdim);
  repack_frag<float><<<nblk((size_t)Hdim * Hdim), 256, 0, stream>>>(rnn_wih, wihR, Hdim, Hdim);
  repack_frag<float><<<nblk((size_t)Hdim * Hdim), 256, 0, stream>>>(rnn_whh, whhR, Hdim, Hdim);

  // ---- layer 1: LSTM ----
  {
    unsigned tiles = (unsigned)((Mrows / 16) * (G4 / 16));
    gemm_bf16<<<tiles / 8, 256, 0, stream>>>(xfrag, wihL, xgbuf, lstm_bih, lstm_bhh,
                                             (int)(Mrows / 16), G4 / 16, Iin / 32, G4);
  }
  scan_init<<<nblk((size_t)Bsz * Hdim), 256, 0, stream>>>(hf0, hf1, cnt);
  lstm_scan<<<16, 256, 131072, stream>>>(xgbuf, whhL, hf0, hf1, Hrow, cnt);

  // ---- layer 2: GRU ----
  repack_frag<__bf16><<<nblk(Mrows * Hdim), 256, 0, stream>>>(Hrow, Hfrag, (int)Mrows, Hdim);
  {
    unsigned tiles = (unsigned)((Mrows / 16) * (G3 / 16));
    gemm_bf16<<<tiles / 8, 256, 0, stream>>>(Hfrag, wihG, xgbuf, gru_bih, nullptr,
                                             (int)(Mrows / 16), G3 / 16, Hdim / 32, G3);
  }
  scan_init<<<nblk((size_t)Bsz * Hdim), 256, 0, stream>>>(hf0, hf1, cnt);
  gru_scan<<<16, 256, 98304, stream>>>(xgbuf, whhG, gru_bhh, hf0, hf1, Hrow, cnt);

  // ---- layer 3: RNN-tanh ----
  repack_frag<__bf16><<<nblk(Mrows * Hdim), 256, 0, stream>>>(Hrow, Hfrag, (int)Mrows, Hdim);
  {
    unsigned tiles = (unsigned)((Mrows / 16) * (Hdim / 16));
    gemm_bf16<<<tiles / 8, 256, 0, stream>>>(Hfrag, wihR, xgbuf, rnn_bih, rnn_bhh,
                                             (int)(Mrows / 16), Hdim / 16, Hdim / 32, Hdim);
  }
  scan_init<<<nblk((size_t)Bsz * Hdim), 256, 0, stream>>>(hf0, hf1, cnt);
  rnn_scan<<<16, 256, 32768, stream>>>(xgbuf, whhR, hf0, hf1, hlast, cnt);

  // ---- FC head ----
  fc_kernel<<<nblk((size_t)Bsz * Odim), 256, 0, stream>>>(hlast, fc_w, fc_b, (float*)d_out);
}
